// LocalAttention_17308718203075
// MI455X (gfx1250) — compile-verified, matrix operand order checked by reading, not run on hardware
//
#include <hip/hip_runtime.h>

#define BB 4
#define NN 8192
#define DD 512        // DIN = DINNER = DOUT
#define WW 128
#define NB (NN / WW)  // 64 windows per batch
#define MTOT (BB * NN)

typedef __attribute__((ext_vector_type(16))) __bf16 v16bf;
typedef __attribute__((ext_vector_type(8)))  __bf16 v8bf;
typedef __attribute__((ext_vector_type(8)))  float  v8f;

// Load a 16(row)x32(K) bf16 tile in the CDNA5 16-bit A/B operand layout:
// lanes 0-15: row = lane,    K chunks {0..7, 16..23}
// lanes16-31: row = lane-16, K chunks {8..15, 24..31}
// (row maps to M for the A operand, N for the B operand.)
__device__ __forceinline__ v16bf load_tile_bf16(const __bf16* base, int ld, int lane) {
  const int r  = lane & 15;
  const int kb = (lane < 16) ? 0 : 8;
  const __bf16* p = base + r * ld + kb;
  v8bf lo = *(const v8bf*)(p);        // K = kb .. kb+7   (16B load)
  v8bf hi = *(const v8bf*)(p + 16);   // K = kb+16 .. kb+23
  v16bf t;
#pragma unroll
  for (int h = 0; h < 8; ++h) { t[h] = lo[h]; t[8 + h] = hi[h]; }
  return t;
}

// Same tile layout, but source is f32 memory: convert to bf16 on the fly.
__device__ __forceinline__ v16bf load_tile_f32(const float* base, int ld, int lane) {
  const int r  = lane & 15;
  const int kb = (lane < 16) ? 0 : 8;
  const float* p = base + r * ld + kb;
  v16bf t;
#pragma unroll
  for (int h = 0; h < 8; ++h) {
    t[h]     = (__bf16)p[h];
    t[8 + h] = (__bf16)p[h + 16];
  }
  return t;
}

// ---------------- Kernel 1: qk = x @ w_qkv[:1024].T  (the V third of w_qkv is
// dead by the reference's replicated v:=k bug, so it is skipped entirely).
// Each wave computes a 32(M) x 64(E) block: 2 A tiles x 4 B tiles -> 8 WMMAs
// per 6 tile loads. Epilogue writes q row-major, k row-major AND k d-major.
__global__ void qk_proj_kernel(const float* __restrict__ x, const float* __restrict__ wqkv,
                               __bf16* __restrict__ qb, __bf16* __restrict__ kb_,
                               __bf16* __restrict__ kT) {
  const int lane = threadIdx.x & 31;
  const int wave = threadIdx.x >> 5;
  const int row0 = blockIdx.x * 256 + wave * 32;  // 8 waves x 32 rows
  const int e0   = blockIdx.y * 64;
  v8f acc[2][4] = {};
#pragma unroll 1
  for (int kt = 0; kt < 16; ++kt) {
    v16bf a0 = load_tile_f32(x + row0 * DD + kt * 32, DD, lane);
    v16bf a1 = load_tile_f32(x + (row0 + 16) * DD + kt * 32, DD, lane);
    if (kt < 14) {  // stream-ahead prefetch of x (w_qkv is L2-resident)
      __builtin_prefetch(x + (row0 + (lane & 15)) * DD + (kt + 2) * 32, 0, 0);
      __builtin_prefetch(x + (row0 + 16 + (lane & 15)) * DD + (kt + 2) * 32, 0, 0);
    }
#pragma unroll
    for (int t = 0; t < 4; ++t) {
      v16bf b = load_tile_f32(wqkv + (e0 + 16 * t) * DD + kt * 32, DD, lane);
      acc[0][t] = __builtin_amdgcn_wmma_f32_16x16x32_bf16(false, a0, false, b,
                                                          (short)0, acc[0][t], false, false);
      acc[1][t] = __builtin_amdgcn_wmma_f32_16x16x32_bf16(false, a1, false, b,
                                                          (short)0, acc[1][t], false, false);
    }
  }
  // C/D layout: reg r, lanes 0-15 -> M=r, lanes 16-31 -> M=r+8; N = lane&15
  const int hi = lane >> 4;
  const int nn = lane & 15;
#pragma unroll
  for (int mi = 0; mi < 2; ++mi) {
#pragma unroll
    for (int t = 0; t < 4; ++t) {
      const int ecol = e0 + 16 * t;
      if (ecol < DD) {  // q columns
#pragma unroll
        for (int r = 0; r < 8; ++r) {
          int m = row0 + 16 * mi + r + 8 * hi;
          qb[m * DD + ecol + nn] = (__bf16)acc[mi][t][r];
        }
      } else {          // k columns: row-major + d-major transpose
        const int ek = ecol - DD + nn;
#pragma unroll
        for (int r = 0; r < 8; ++r) {
          int m = row0 + 16 * mi + r + 8 * hi;
          __bf16 v = (__bf16)acc[mi][t][r];
          kb_[m * DD + ek] = v;
          int bidx = m >> 13, n = m & (NN - 1);
          kT[(bidx * DD + ek) * NN + n] = v;
        }
      }
    }
  }
}

// ---------------- Kernel 2: fused local attention for one (batch, window).
// Wave w owns a 16-column strip of simT = k2 @ q^T (256 x 16), does the causal
// mask + softmax over j in registers, packs attn^T directly into WMMA B operands
// (pure in-lane repack thanks to the simT orientation), then outT = k2^T @ attn^T.
__global__ void local_attn_kernel(const __bf16* __restrict__ qb, const __bf16* __restrict__ kb_,
                                  const __bf16* __restrict__ kT, __bf16* __restrict__ ob) {
  const int lane = threadIdx.x & 31;
  const int wave = threadIdx.x >> 5;
  const int wb = blockIdx.x;   // window index
  const int b  = blockIdx.y;   // batch
  const float scale = 0.04419417382415922f;  // 512^-0.5
  const int hi = lane >> 4;

  // Hoist the wave's q strip (16 rows x 512) as 16 B-operand tiles.
  const int qrow = b * NN + wb * WW + wave * 16;
  v16bf qt[16];
#pragma unroll
  for (int kt = 0; kt < 16; ++kt)
    qt[kt] = load_tile_bf16(qb + qrow * DD + kt * 32, DD, lane);

  // simT tiles: jt = 0..15 covers j = 0..255 (prev window + current window).
  // j < 128 rows come from window wb-1; for wb==0 those are the zero pad bucket
  // (reference keeps them unmasked with sim = 0), so we just skip the matmuls.
  const int krow0 = b * NN + (wb - 1) * WW;
  const int iwin  = wave * 16 + (lane & 15);  // q row within window (0..127)
  v8f simT[16];
#pragma unroll 1
  for (int jt = 0; jt < 16; ++jt) {
    v8f acc = {};
    if (!(wb == 0 && jt < 8)) {
#pragma unroll
      for (int kt = 0; kt < 16; ++kt) {
        v16bf a = load_tile_bf16(kb_ + (krow0 + jt * 16) * DD + kt * 32, DD, lane);
        acc = __builtin_amdgcn_wmma_f32_16x16x32_bf16(false, a, false, qt[kt],
                                                      (short)0, acc, false, false);
      }
    }
    // scale + causal mask: masked iff j > i + 128 (matches triu(.., 129))
#pragma unroll
    for (int r = 0; r < 8; ++r) {
      int j = jt * 16 + r + 8 * hi;
      float s = acc[r] * scale;
      if (j > iwin + WW) s = -3.402823466e38f;
      acc[r] = s;
    }
    simT[jt] = acc;
  }

  // Softmax over j for each column i. i is fixed per lane (= lane&15); the two
  // lane half-groups hold disjoint j's of the same column -> one xor-16 shuffle.
  float mx = -3.402823466e38f;
#pragma unroll
  for (int jt = 0; jt < 16; ++jt)
#pragma unroll
    for (int r = 0; r < 8; ++r) mx = fmaxf(mx, simT[jt][r]);
  mx = fmaxf(mx, __shfl_xor(mx, 16, 32));
  float sum = 0.f;
#pragma unroll
  for (int jt = 0; jt < 16; ++jt)
#pragma unroll
    for (int r = 0; r < 8; ++r) {
      float p = __expf(simT[jt][r] - mx);
      simT[jt][r] = p;
      sum += p;
    }
  sum += __shfl_xor(sum, 16, 32);
  const float inv = 1.0f / sum;

  // Pack attn^T as B operands: chunk c covers K(=j) 32c..32c+31.
  // In-lane repack: halfs 0..7 <- simT[2c] regs, halfs 8..15 <- simT[2c+1] regs.
  v16bf attnB[8];
#pragma unroll
  for (int c = 0; c < 8; ++c) {
    v16bf t;
#pragma unroll
    for (int h = 0; h < 8; ++h) {
      t[h]     = (__bf16)(simT[2 * c][h]     * inv);
      t[8 + h] = (__bf16)(simT[2 * c + 1][h] * inv);
    }
    attnB[c] = t;
  }

  // outT = k2^T @ attn^T. A tiles from the d-major kT copy (contiguous 16B loads).
  // v2 == k2 per the reference's replicated-source bug.
  // Epilogue: for a fixed lane, regs r=0..7 are consecutive d -> one 16B store.
  const int nout = wb * WW + wave * 16 + (lane & 15);  // seq row within batch
  const int kTn0 = (wb - 1) * WW;
  const int c0   = (wb == 0) ? 4 : 0;  // zero pad bucket contributes nothing
  __bf16* obrow = ob + (b * NN + nout) * DD;
#pragma unroll 1
  for (int dt = 0; dt < 32; ++dt) {
    v8f acc = {};
    for (int c = c0; c < 8; ++c) {
      v16bf a = load_tile_bf16(kT + (b * DD + dt * 16) * NN + kTn0 + 32 * c, NN, lane);
      acc = __builtin_amdgcn_wmma_f32_16x16x32_bf16(false, a, false, attnB[c],
                                                    (short)0, acc, false, false);
    }
    v8bf o;
#pragma unroll
    for (int r = 0; r < 8; ++r) o[r] = (__bf16)acc[r];
    *(v8bf*)(obrow + dt * 16 + 8 * hi) = o;   // d = dt*16 + 8*hi + (0..7)
  }
}

// ---------------- Kernel 3: final = out @ w_out^T + b_out (f32 result).
// Same 32x64-per-wave register blocking as kernel 1.
__global__ void out_proj_kernel(const __bf16* __restrict__ ob, const float* __restrict__ wout,
                                const float* __restrict__ bout, float* __restrict__ out) {
  const int lane = threadIdx.x & 31;
  const int wave = threadIdx.x >> 5;
  const int row0 = blockIdx.x * 256 + wave * 32;
  const int o0   = blockIdx.y * 64;
  v8f acc[2][4] = {};
#pragma unroll 1
  for (int kt = 0; kt < 16; ++kt) {
    v16bf a0 = load_tile_bf16(ob + row0 * DD + kt * 32, DD, lane);
    v16bf a1 = load_tile_bf16(ob + (row0 + 16) * DD + kt * 32, DD, lane);
#pragma unroll
    for (int t = 0; t < 4; ++t) {
      v16bf b = load_tile_f32(wout + (o0 + 16 * t) * DD + kt * 32, DD, lane);
      acc[0][t] = __builtin_amdgcn_wmma_f32_16x16x32_bf16(false, a0, false, b,
                                                          (short)0, acc[0][t], false, false);
      acc[1][t] = __builtin_amdgcn_wmma_f32_16x16x32_bf16(false, a1, false, b,
                                                          (short)0, acc[1][t], false, false);
    }
  }
  const int hi = lane >> 4, nn = lane & 15;
#pragma unroll
  for (int mi = 0; mi < 2; ++mi) {
#pragma unroll
    for (int t = 0; t < 4; ++t) {
      int o = o0 + 16 * t + nn;
      float bb = bout[o];
#pragma unroll
      for (int r = 0; r < 8; ++r) {
        int m = row0 + 16 * mi + r + 8 * hi;
        out[m * DD + o] = acc[mi][t][r] + bb;
      }
    }
  }
}

extern "C" void kernel_launch(void* const* d_in, const int* in_sizes, int n_in,
                              void* d_out, int out_size, void* d_ws, size_t ws_size,
                              hipStream_t stream) {
  const float* x    = (const float*)d_in[0];  // (4, 8192, 512)
  const float* wqkv = (const float*)d_in[1];  // (1536, 512)
  const float* wout = (const float*)d_in[2];  // (512, 512)
  const float* bout = (const float*)d_in[3];  // (512,)
  float* out = (float*)d_out;

  char* ws = (char*)d_ws;
  const size_t SZ = (size_t)MTOT * DD * sizeof(__bf16);  // 32 MB each
  __bf16* qb  = (__bf16*)(ws);           // q,  row-major [B*N][D]
  __bf16* kb_ = (__bf16*)(ws + SZ);      // k,  row-major [B*N][D]
  __bf16* kT  = (__bf16*)(ws + 2 * SZ);  // k,  d-major   [B][D][N]
  __bf16* ob  = (__bf16*)(ws + 3 * SZ);  // attn out, row-major [B*N][D]

  dim3 blk(256);  // 8 waves (wave32)
  qk_proj_kernel<<<dim3(MTOT / 256, 1024 / 64), blk, 0, stream>>>(x, wqkv, qb, kb_, kT);
  local_attn_kernel<<<dim3(NB, BB), blk, 0, stream>>>(qb, kb_, kT, ob);
  out_proj_kernel<<<dim3(MTOT / 256, DD / 64), blk, 0, stream>>>(ob, wout, bout, out);
}
